// PaperGMRF_79207786873398
// MI455X (gfx1250) — compile-verified, matrix operand order checked by reading, not run on hardware
//
#include <hip/hip_runtime.h>
#include <hip/hip_bf16.h>
#include <math.h>

typedef _Float16 h16;
typedef __attribute__((ext_vector_type(16))) _Float16 v16h;
typedef __attribute__((ext_vector_type(8)))  _Float16 v8h;
typedef __attribute__((ext_vector_type(8)))  float    v8f;
typedef int v4i_vs __attribute__((vector_size(4 * sizeof(int))));

#define NN   50000
#define DEGC 16
#define EW   (NN*DEGC)

#define BM 128
#define BN 128
#define BK 32
#define LDA 40   // padded LDS row stride in f16 (80B -> no 16-lane bank conflicts)

static __device__ __forceinline__ v16h mk16(v8h lo, v8h hi) {
  v16h r;
#pragma unroll
  for (int i = 0; i < 8; ++i) { r[i] = lo[i]; r[i + 8] = hi[i]; }
  return r;
}

// ---- CDNA5 async global->LDS copy (ASYNCcnt-tracked), 16B per lane ----
#if __has_builtin(__builtin_amdgcn_global_load_async_to_lds_b128)
#define HAVE_ASYNC_LDS 1
#else
#define HAVE_ASYNC_LDS 0
#endif

static __device__ __forceinline__ void cp_b128(h16* l, const h16* g) {
#if HAVE_ASYNC_LDS
  __builtin_amdgcn_global_load_async_to_lds_b128(
      (__attribute__((address_space(1))) v4i_vs*)(g),
      (__attribute__((address_space(3))) v4i_vs*)(l), 0, 0);
#else
  *(uint4*)l = *(const uint4*)g;
#endif
}

static __device__ __forceinline__ void wait_async_done() {
#if HAVE_ASYNC_LDS
  asm volatile("s_wait_asynccnt 0x0" ::: "memory");
#else
  asm volatile("s_wait_idle" ::: "memory");
#endif
}

// ---------------------------------------------------------------------------
// f16-input WMMA GEMM, async double-buffered LDS staging for A and B tiles.
// C[M,Nc] = act(A[M,K] * Bt^T + bias), Bt pre-transposed [Nc,K] f16.
// ---------------------------------------------------------------------------
template <int ACT, int OUT16>
__global__ __launch_bounds__(256) void gemm_f16_async_kernel(
    const h16* __restrict__ Ap, const h16* __restrict__ Bt,
    const float* __restrict__ bias, void* __restrict__ Cp,
    int M, int K, int Nc)
{
  __shared__ alignas(16) h16 sA[2][BM * LDA];
  __shared__ alignas(16) h16 sB[2][BN * LDA];

  const int tid  = threadIdx.x;
  const int wave = tid >> 5;
  const int lane = tid & 31;
  const int m0 = blockIdx.x * BM;
  const int n0 = blockIdx.y * BN;
  const int wm = (wave & 3) * 32;   // 4 waves tile M
  const int wn = (wave >> 2) * 64;  // 2 waves tile N
  const int ln = lane & 15;
  const int kb = (lane >> 4) * 8;   // k-half select per WMMA 16-bit layout

  // per-thread tile-load slot: 2 threads per row, 8 f16 (16B) each
  const int r   = tid >> 1;
  const int seg = (tid & 1) * 16;
  int gma = m0 + r; if (gma >= M) gma = M - 1;        // clamp; OOB rows never stored
  const h16* gA = Ap + (size_t)gma * K + seg;
  const h16* gB = Bt + (size_t)(n0 + r) * K + seg;
  const int ldslot = r * LDA + seg;

  v8f acc[2][4];
#pragma unroll
  for (int i = 0; i < 2; ++i)
#pragma unroll
    for (int j = 0; j < 4; ++j)
      acc[i][j] = (v8f){0.f, 0.f, 0.f, 0.f, 0.f, 0.f, 0.f, 0.f};

  const int T = K / BK;
  // prologue: tile 0 in flight
  cp_b128(&sA[0][ldslot], gA);
  cp_b128(&sB[0][ldslot], gB);

  for (int t = 0; t < T; ++t) {
    wait_async_done();     // own tile-t chunks landed in LDS
    __syncthreads();       // everyone's landed; everyone done reading tile t-1
    const int cur = t & 1;
    if (t + 1 < T) {       // overlap next tile's DMA with this tile's WMMAs
      const int k1 = (t + 1) * BK;
      cp_b128(&sA[cur ^ 1][ldslot], gA + k1);
      cp_b128(&sB[cur ^ 1][ldslot], gB + k1);
    }

    v16h bf[4];
#pragma unroll
    for (int j = 0; j < 4; ++j) {
      const h16* bp = &sB[cur][(wn + j * 16 + ln) * LDA + kb];
      bf[j] = mk16(*(const v8h*)bp, *(const v8h*)(bp + 16));
    }
#pragma unroll
    for (int i = 0; i < 2; ++i) {
      const h16* ap = &sA[cur][(wm + i * 16 + ln) * LDA + kb];
      v16h af = mk16(*(const v8h*)ap, *(const v8h*)(ap + 16));
#pragma unroll
      for (int j = 0; j < 4; ++j)
        acc[i][j] = __builtin_amdgcn_wmma_f32_16x16x32_f16(
            false, af, false, bf[j], (short)0, acc[i][j], false, false);
    }
  }

  // ---- epilogue: bias + activation + store ----
#pragma unroll
  for (int i = 0; i < 2; ++i) {
#pragma unroll
    for (int j = 0; j < 4; ++j) {
      const int n  = n0 + wn + j * 16 + ln;
      const float bv = bias[n];
      const int mb = m0 + wm + i * 16 + (lane >> 4) * 8;
#pragma unroll
      for (int v = 0; v < 8; ++v) {
        const int m = mb + v;
        if (m < M) {
          float x = acc[i][j][v] + bv;
          if (ACT) x = x > 0.f ? x : 0.f;
          if (OUT16) ((h16*)Cp)[(size_t)m * Nc + n] = (h16)x;
          else       ((float*)Cp)[(size_t)m * Nc + n] = x;
        }
      }
    }
  }
}

// ---------------------------------------------------------------------------
// f32-input first-layer GEMM (converts A to f16 while staging to LDS).
// Always ReLU + f16 output. B fragments straight from L2.
// ---------------------------------------------------------------------------
__global__ __launch_bounds__(256) void gemm_f32in_kernel(
    const float* __restrict__ Ap, const h16* __restrict__ Bt,
    const float* __restrict__ bias, h16* __restrict__ Cp,
    int M, int K, int Nc)
{
  __shared__ alignas(16) h16 sA[BM * LDA];

  const int tid  = threadIdx.x;
  const int wave = tid >> 5;
  const int lane = tid & 31;
  const int m0 = blockIdx.x * BM;
  const int n0 = blockIdx.y * BN;
  const int wm = (wave & 3) * 32;
  const int wn = (wave >> 2) * 64;
  const int ln = lane & 15;
  const int kb = (lane >> 4) * 8;

  v8f acc[2][4];
#pragma unroll
  for (int i = 0; i < 2; ++i)
#pragma unroll
    for (int j = 0; j < 4; ++j)
      acc[i][j] = (v8f){0.f, 0.f, 0.f, 0.f, 0.f, 0.f, 0.f, 0.f};

  for (int k0 = 0; k0 < K; k0 += BK) {
    __syncthreads();
    {
      const int r   = tid >> 1;
      const int seg = (tid & 1) * 16;
      const int gm  = m0 + r;
      float f[16];
#pragma unroll
      for (int t = 0; t < 16; ++t) f[t] = 0.f;
      if (gm < M) {
        const float4* p = (const float4*)(Ap + (size_t)gm * K + k0 + seg);
#pragma unroll
        for (int q = 0; q < 4; ++q) {
          float4 v = p[q];
          f[q * 4 + 0] = v.x; f[q * 4 + 1] = v.y; f[q * 4 + 2] = v.z; f[q * 4 + 3] = v.w;
        }
      }
      v8h h0, h1;
#pragma unroll
      for (int t = 0; t < 8; ++t) { h0[t] = (h16)f[t]; h1[t] = (h16)f[8 + t]; }
      *(v8h*)&sA[r * LDA + seg]     = h0;
      *(v8h*)&sA[r * LDA + seg + 8] = h1;
    }
    __syncthreads();

    v16h bf[4];
#pragma unroll
    for (int j = 0; j < 4; ++j) {
      const int n = n0 + wn + j * 16 + ln;
      const h16* bp = Bt + (size_t)n * K + k0 + kb;
      bf[j] = mk16(*(const v8h*)bp, *(const v8h*)(bp + 16));
    }
#pragma unroll
    for (int i = 0; i < 2; ++i) {
      const h16* ap = &sA[(wm + i * 16 + ln) * LDA + kb];
      v16h af = mk16(*(const v8h*)ap, *(const v8h*)(ap + 16));
#pragma unroll
      for (int j = 0; j < 4; ++j)
        acc[i][j] = __builtin_amdgcn_wmma_f32_16x16x32_f16(
            false, af, false, bf[j], (short)0, acc[i][j], false, false);
    }
  }

#pragma unroll
  for (int i = 0; i < 2; ++i) {
#pragma unroll
    for (int j = 0; j < 4; ++j) {
      const int n  = n0 + wn + j * 16 + ln;
      const float bv = bias[n];
      const int mb = m0 + wm + i * 16 + (lane >> 4) * 8;
#pragma unroll
      for (int v = 0; v < 8; ++v) {
        const int m = mb + v;
        if (m < M) {
          float x = acc[i][j][v] + bv;
          x = x > 0.f ? x : 0.f;
          Cp[(size_t)m * Nc + n] = (h16)x;
        }
      }
    }
  }
}

// ---------------------------------------------------------------------------
// Convert/transpose weights: W[K,Nc] f32 -> Wt[Nc,K] f16
// ---------------------------------------------------------------------------
__global__ void wconv_kernel(const float* __restrict__ W, h16* __restrict__ Wt,
                             int K, int Nc)
{
  int idx = blockIdx.x * blockDim.x + threadIdx.x;
  if (idx >= K * Nc) return;
  int k = idx / Nc, n = idx - k * Nc;
  Wt[(size_t)n * K + k] = (h16)W[idx];
}

// wbu = Wb @ ab[:64] etc. (GAT scores collapse: no nonlinearity between projections)
__global__ void gatvec_kernel(const float* __restrict__ Wb, const float* __restrict__ ab,
                              const float* __restrict__ Wp, const float* __restrict__ ap,
                              float* wbu, float* wbv, float* wpu, float* wpv)
{
  int j = threadIdx.x;  // 128 threads
  float s1 = 0.f, s2 = 0.f, s3 = 0.f, s4 = 0.f;
#pragma unroll 4
  for (int g = 0; g < 64; ++g) {
    float b = Wb[j * 64 + g], p = Wp[j * 64 + g];
    s1 += b * ab[g];      s2 += b * ab[64 + g];
    s3 += p * ap[g];      s4 += p * ap[64 + g];
  }
  wbu[j] = s1; wbv[j] = s2; wpu[j] = s3; wpv[j] = s4;
}

// Per-node: phi (sigmoid of node potential) + 4 GAT dot products.
__global__ void node_kernel(const float* __restrict__ zb, const float* __restrict__ zp,
                            const float* __restrict__ nodeW, const float* __restrict__ nodeb,
                            const float* __restrict__ wbu, const float* __restrict__ wbv,
                            const float* __restrict__ wpu, const float* __restrict__ wpv,
                            float* __restrict__ phi, float* __restrict__ ub, float* __restrict__ vb,
                            float* __restrict__ up, float* __restrict__ vp)
{
  int n = blockIdx.x * blockDim.x + threadIdx.x;
  if (n >= NN) return;
  const float* a = zb + (size_t)n * 128;
  const float* b = zp + (size_t)n * 128;
  float d0 = 0.f, d1 = 0.f, d2 = 0.f, d3 = 0.f, d4 = 0.f;
#pragma unroll 4
  for (int j = 0; j < 128; ++j) {
    float x = a[j], y = b[j];
    d0 += (x + y) * nodeW[j];
    d1 += x * wbu[j]; d2 += x * wbv[j];
    d3 += y * wpu[j]; d4 += y * wpv[j];
  }
  float pr = 1.f / (1.f + __expf(-(d0 + nodeb[0])));
  phi[2 * n] = 1.f - pr; phi[2 * n + 1] = pr;
  ub[n] = d1; vb[n] = d2; up[n] = d3; vp[n] = d4;
}

static __device__ __forceinline__ float lrelu(float v) { return v >= 0.f ? v : 0.2f * v; }

// Per undirected edge: Potts diagonal weight exp(sigmoid(w0*sb + w1*sp))
__global__ void edge_kernel(const int* __restrict__ ei,
                            const float* __restrict__ ub, const float* __restrict__ vb,
                            const float* __restrict__ up, const float* __restrict__ vp,
                            const float* __restrict__ gamma, float* __restrict__ same_e)
{
  int e = blockIdx.x * blockDim.x + threadIdx.x;
  if (e >= EW) return;
  int i = ei[e], j = ei[EW + e];
  float sb = 0.5f * (lrelu(ub[i] + vb[j]) + lrelu(ub[j] + vb[i]));
  float sp = 0.5f * (lrelu(up[i] + vp[j]) + lrelu(up[j] + vp[i]));
  float g0 = __expf(gamma[0]), g1 = __expf(gamma[1]);
  float s = (g0 * sb + g1 * sp) / (g0 + g1);   // TAU = 1
  float alpha = 1.f / (1.f + __expf(-s));
  same_e[e] = __expf(alpha);                   // BETA = 1
}

__global__ void init_msgs_kernel(float* __restrict__ m, int n)
{
  int i = blockIdx.x * blockDim.x + threadIdx.x;
  if (i < n) m[i] = 1.0f;
}

// node_prod[i] = phi[i] * prod over 32 incoming messages (graph is (i, i+k+1 mod N))
__global__ void nodeprod_kernel(const float* __restrict__ msgs, const float* __restrict__ phi,
                                float* __restrict__ nprod)
{
  int i = blockIdx.x * blockDim.x + threadIdx.x;
  if (i >= NN) return;
  float p0 = phi[2 * i], p1 = phi[2 * i + 1];
#pragma unroll
  for (int k = 0; k < DEGC; ++k) {
    int i0 = i - k - 1; if (i0 < 0) i0 += NN;
    int d1 = k * NN + i0;        // first half: i0 -> i
    int d2 = EW + k * NN + i;    // second half: (i+k+1) -> i
    p0 *= msgs[2 * d1]     * msgs[2 * d2];
    p1 *= msgs[2 * d1 + 1] * msgs[2 * d2 + 1];
  }
  nprod[2 * i] = p0; nprod[2 * i + 1] = p1;
}

// Synchronous message update with leave-one-out division, then normalize.
__global__ void msg_kernel(const float* __restrict__ msgs, const float* __restrict__ nprod,
                           const float* __restrict__ same_e, const int* __restrict__ ei,
                           float* __restrict__ out)
{
  int d = blockIdx.x * blockDim.x + threadIdx.x;
  if (d >= 2 * EW) return;
  int e   = d < EW ? d : d - EW;
  int src = d < EW ? ei[e] : ei[EW + e];
  int rev = d < EW ? d + EW : d - EW;
  float r0 = msgs[2 * rev], r1 = msgs[2 * rev + 1];
  float q0 = nprod[2 * src] / r0;
  float q1 = nprod[2 * src + 1] / r1;
  float s = same_e[e];
  float m0 = s * q0 + q1;
  float m1 = q0 + s * q1;
  float inv = 1.f / (m0 + m1 + 1e-8f);
  out[2 * d] = m0 * inv;
  out[2 * d + 1] = m1 * inv;
}

__global__ void belief_kernel(const float* __restrict__ nprod, float* __restrict__ out)
{
  int i = blockIdx.x * blockDim.x + threadIdx.x;
  if (i >= NN) return;
  float p0 = nprod[2 * i], p1 = nprod[2 * i + 1];
  out[i] = p1 / (p0 + p1 + 1e-8f);
}

// ---------------------------------------------------------------------------

extern "C" void kernel_launch(void* const* d_in, const int* in_sizes, int n_in,
                              void* d_out, int out_size, void* d_ws, size_t ws_size,
                              hipStream_t stream)
{
  (void)in_sizes; (void)n_in; (void)out_size; (void)ws_size;

  const float* x_beh  = (const float*)d_in[0];
  const float* x_pre  = (const float*)d_in[1];
  const int*   ei     = (const int*)d_in[2];
  const float* beh1_W = (const float*)d_in[3];  const float* beh1_b = (const float*)d_in[4];
  const float* beh2_W = (const float*)d_in[5];  const float* beh2_b = (const float*)d_in[6];
  const float* pre1_W = (const float*)d_in[7];  const float* pre1_b = (const float*)d_in[8];
  const float* pre2_W = (const float*)d_in[9];  const float* pre2_b = (const float*)d_in[10];
  const float* sh1_W  = (const float*)d_in[11]; const float* sh1_b  = (const float*)d_in[12];
  const float* sh2_W  = (const float*)d_in[13]; const float* sh2_b  = (const float*)d_in[14];
  const float* node_W = (const float*)d_in[15]; const float* node_b = (const float*)d_in[16];
  const float* Wb     = (const float*)d_in[17];
  const float* Wp     = (const float*)d_in[18];
  const float* ab     = (const float*)d_in[19];
  const float* ap     = (const float*)d_in[20];
  const float* gamma  = (const float*)d_in[21];

  // ---- workspace carve ----
  char* wp_ = (char*)d_ws;
  auto carve = [&](size_t bytes) -> void* {
    void* r = (void*)wp_;
    wp_ += (bytes + 255) & ~(size_t)255;
    return r;
  };
  h16* W1bT = (h16*)carve((size_t)256 * 512 * sizeof(h16));
  h16* W2bT = (h16*)carve((size_t)512 * 512 * sizeof(h16));
  h16* W1pT = (h16*)carve((size_t)256 * 512 * sizeof(h16));
  h16* W2pT = (h16*)carve((size_t)512 * 512 * sizeof(h16));
  h16* Ws1T = (h16*)carve((size_t)512 * 512 * sizeof(h16));
  h16* Ws2T = (h16*)carve((size_t)512 * 128 * sizeof(h16));
  float* wbu = (float*)carve(128 * sizeof(float));
  float* wbv = (float*)carve(128 * sizeof(float));
  float* wpu = (float*)carve(128 * sizeof(float));
  float* wpv = (float*)carve(128 * sizeof(float));
  float* Zb  = (float*)carve((size_t)NN * 128 * sizeof(float));
  float* Zp  = (float*)carve((size_t)NN * 128 * sizeof(float));
  float* phi = (float*)carve((size_t)NN * 2 * sizeof(float));
  float* ubuf  = (float*)carve((size_t)NN * sizeof(float));
  float* vbuf  = (float*)carve((size_t)NN * sizeof(float));
  float* upbuf = (float*)carve((size_t)NN * sizeof(float));
  float* vpbuf = (float*)carve((size_t)NN * sizeof(float));
  float* same_e = (float*)carve((size_t)EW * sizeof(float));
  float* nprod  = (float*)carve((size_t)NN * 2 * sizeof(float));
  h16* T1 = (h16*)carve((size_t)NN * 512 * sizeof(h16));   // reused as msgs buf A
  h16* T2 = (h16*)carve((size_t)NN * 512 * sizeof(h16));   // reused as msgs buf B

  const dim3 blk(256);
  auto g1 = [](int n) { return dim3((unsigned)((n + 255) / 256)); };

  // ---- weight prep ----
  wconv_kernel<<<g1(256 * 512), blk, 0, stream>>>(beh1_W, W1bT, 256, 512);
  wconv_kernel<<<g1(512 * 512), blk, 0, stream>>>(beh2_W, W2bT, 512, 512);
  wconv_kernel<<<g1(256 * 512), blk, 0, stream>>>(pre1_W, W1pT, 256, 512);
  wconv_kernel<<<g1(512 * 512), blk, 0, stream>>>(pre2_W, W2pT, 512, 512);
  wconv_kernel<<<g1(512 * 512), blk, 0, stream>>>(sh1_W,  Ws1T, 512, 512);
  wconv_kernel<<<g1(512 * 128), blk, 0, stream>>>(sh2_W,  Ws2T, 512, 128);
  gatvec_kernel<<<dim3(1), dim3(128), 0, stream>>>(Wb, ab, Wp, ap, wbu, wbv, wpu, wpv);

  // ---- GEMM chain (T1/T2 ping-pong) ----
  const unsigned gMx = (NN + BM - 1) / BM;
  dim3 grid512(gMx, 512 / BN), grid128(gMx, 128 / BN);

  gemm_f32in_kernel<<<grid512, blk, 0, stream>>>(x_beh, W1bT, beh1_b, T1, NN, 256, 512);
  gemm_f16_async_kernel<0, 1><<<grid512, blk, 0, stream>>>(T1, W2bT, beh2_b, T2, NN, 512, 512);
  gemm_f16_async_kernel<1, 1><<<grid512, blk, 0, stream>>>(T2, Ws1T, sh1_b, T1, NN, 512, 512);
  gemm_f16_async_kernel<0, 0><<<grid128, blk, 0, stream>>>(T1, Ws2T, sh2_b, Zb, NN, 512, 128);

  gemm_f32in_kernel<<<grid512, blk, 0, stream>>>(x_pre, W1pT, pre1_b, T2, NN, 256, 512);
  gemm_f16_async_kernel<0, 1><<<grid512, blk, 0, stream>>>(T2, W2pT, pre2_b, T1, NN, 512, 512);
  gemm_f16_async_kernel<1, 1><<<grid512, blk, 0, stream>>>(T1, Ws1T, sh1_b, T2, NN, 512, 512);
  gemm_f16_async_kernel<0, 0><<<grid128, blk, 0, stream>>>(T2, Ws2T, sh2_b, Zp, NN, 512, 128);

  // ---- node / edge potentials ----
  node_kernel<<<g1(NN), blk, 0, stream>>>(Zb, Zp, node_W, node_b, wbu, wbv, wpu, wpv,
                                          phi, ubuf, vbuf, upbuf, vpbuf);
  edge_kernel<<<g1(EW), blk, 0, stream>>>(ei, ubuf, vbuf, upbuf, vpbuf, gamma, same_e);

  // ---- loopy BP (messages alias T1/T2, which are now dead) ----
  float* msgs_a = (float*)T1;
  float* msgs_b = (float*)T2;
  init_msgs_kernel<<<g1(2 * EW * 2), blk, 0, stream>>>(msgs_a, 2 * EW * 2);
  for (int it = 0; it < 10; ++it) {
    nodeprod_kernel<<<g1(NN), blk, 0, stream>>>(msgs_a, phi, nprod);
    msg_kernel<<<g1(2 * EW), blk, 0, stream>>>(msgs_a, nprod, same_e, ei, msgs_b);
    float* t = msgs_a; msgs_a = msgs_b; msgs_b = t;
  }
  nodeprod_kernel<<<g1(NN), blk, 0, stream>>>(msgs_a, phi, nprod);
  belief_kernel<<<g1(NN), blk, 0, stream>>>(nprod, (float*)d_out);
}
